// ConvMotion_85323820302552
// MI455X (gfx1250) — matrix operand order repaired and checked by reading, not compiled, not run on hardware
//
#include <hip/hip_runtime.h>
#include <hip/hip_bf16.h>

// ---------------------------------------------------------------------------
// MI455X plan (round 2):
//  * Conv branch -> v_wmma_f32_16x16x32_bf16, bf16x2 split (hi*hi+hi*lo+lo*hi,
//    f32 accum). K = ic = 32 matches WMMA K exactly.
//  * Round-2 change: each wave owns ONE oc-tile and FOUR 16-pixel N-tiles, so
//    each A (weight) fragment load is reused by 4 WMMA triplets. This cuts the
//    redundant L2 weight-fragment traffic 4x (was the dominant vmem stream)
//    while keeping the same WMMA count per pixel and lower register pressure.
//  * x tile staged in LDS as bf16 hi/lo, ic-contiguous per (row,col), 144B
//    chunk stride; B fragment per lane = two aligned ds_load_b128.
//  * Motion branch: gather/VALU kernel; coords+weights computed once per
//    pixel, reused across all 32 channels; x stays hot in the 192MB L2.
// ---------------------------------------------------------------------------

typedef __bf16 bf16_t;
typedef __attribute__((ext_vector_type(16))) __bf16 v16bf;
typedef __attribute__((ext_vector_type(8)))  __bf16 v8bf;
typedef __attribute__((ext_vector_type(8)))  float  v8f;

#define HW    65536      // 256*256
#define WDIM  256
#define HDIM  256
#define CIN   32
#define TILE_COLS 68     // 64 outputs + 4 halo
#define CHUNK 144        // bytes per (row,col): 64B hi + 64B lo + 16B pad

__device__ __forceinline__ v16bf load16(const bf16_t* p) {
  v8bf a = *(const v8bf*)p;
  v8bf b = *(const v8bf*)(p + 8);
  v16bf r;
#pragma unroll
  for (int i = 0; i < 8; ++i) { r[i] = a[i]; r[i + 8] = b[i]; }
  return r;
}

// ---------------------------------------------------------------------------
// Pre-swizzle weights into WMMA A-fragment order (bf16 hi + residual lo).
// A order used here (contiguous k, same as B): lane l (m = l&15), value j in 0..15 maps to
//   k = j + 16*(l>=16)
// Entry index: ((tap*2 + ocTile)*32 + lane)*16 + j
// ---------------------------------------------------------------------------
__global__ void PrepWeightFrag(const float* __restrict__ w,
                               bf16_t* __restrict__ whi,
                               bf16_t* __restrict__ wlo) {
  int idx = blockIdx.x * 256 + threadIdx.x;      // 25*2*32*16 = 25600 entries
  if (idx >= 25 * 2 * 32 * 16) return;
  int j = idx & 15;
  int l = (idx >> 4) & 31;
  int o = (idx >> 9) & 1;
  int t = idx >> 10;                             // tap = ky*5 + kx
  int m = l & 15;
  int k = j + ((l >= 16) ? 16 : 0);
  int oc = o * 16 + m;
  int ky = t / 5, kx = t % 5;
  float v = w[((oc * CIN + k) * 5 + ky) * 5 + kx];
  bf16_t hi = (bf16_t)v;
  float rem = v - (float)hi;
  whi[idx] = hi;
  wlo[idx] = (bf16_t)rem;
}

// ---------------------------------------------------------------------------
// Conv branch: one 64-thread block (2 waves) = 64 output pixels of one row.
// Wave w owns oc-tile w (16 oc) and all four 16-pixel N-tiles:
//   per tap: 1 A hi/lo fragment load, reused across 4 N-tiles -> 12 WMMAs.
// 25 taps * 12 = 300 WMMAs per wave.
// ---------------------------------------------------------------------------
__global__ __launch_bounds__(64) void ConvWmma(const float* __restrict__ x,
                                               const bf16_t* __restrict__ whi,
                                               const bf16_t* __restrict__ wlo,
                                               float* __restrict__ out) {
  __shared__ __align__(16) unsigned char lds[5 * TILE_COLS * CHUNK];

  const int x0  = blockIdx.x * 64;   // 0,64,128,192
  const int y   = blockIdx.y;        // row
  const int b   = blockIdx.z;        // batch
  const int tid = threadIdx.x;
  const float* xb = x + (size_t)b * CIN * HW;

  // Stage halo tile rows y-2..y+2, cols x0-2..x0+65, all 32 ic, as bf16 hi/lo.
  // col is the fastest index across consecutive threads -> coalesced global
  // loads (replicate-clamped at the borders).
  for (int idx = tid; idx < 5 * TILE_COLS * CIN; idx += 64) {
    int col  = idx % TILE_COLS;
    int rest = idx / TILE_COLS;
    int row  = rest % 5;
    int ic   = rest / 5;
    int gy = y + row - 2;  gy = gy < 0 ? 0 : (gy > HDIM - 1 ? HDIM - 1 : gy);
    int gx = x0 + col - 2; gx = gx < 0 ? 0 : (gx > WDIM - 1 ? WDIM - 1 : gx);
    float v = xb[(ic * HDIM + gy) * WDIM + gx];
    bf16_t hi = (bf16_t)v;
    bf16_t lo = (bf16_t)(v - (float)hi);
    bf16_t* c = (bf16_t*)(lds + (row * TILE_COLS + col) * CHUNK);
    c[ic]      = hi;   // hi block: ic 0..31
    c[32 + ic] = lo;   // lo block: ic 0..31
  }
  __syncthreads();

  const int lane  = tid & 31;
  const int oc    = tid >> 5;        // wave 0 -> oc 0..15, wave 1 -> oc 16..31
  const int n     = lane & 15;       // N = pixel within 16-px tile
  const int khalf = lane >> 4;       // B as loaded here: lanes 16-31 hold K=16..31

  v8f acc[4] = {v8f{}, v8f{}, v8f{}, v8f{}};   // 4 N-tiles (64 pixels)

#pragma unroll
  for (int r = 0; r < 5; ++r) {
#pragma unroll
    for (int kx = 0; kx < 5; ++kx) {
      const int t = r * 5 + kx;
      // One oc-tile's weight fragment, reused by all 4 N-tiles below.
      v16bf Ah = load16(whi + ((size_t)(t * 2 + oc) * 32 + lane) * 16);
      v16bf Al = load16(wlo + ((size_t)(t * 2 + oc) * 32 + lane) * 16);
#pragma unroll
      for (int s = 0; s < 4; ++s) {
        const bf16_t* bp =
            (const bf16_t*)(lds + (r * TILE_COLS + s * 16 + n + kx) * CHUNK);
        v16bf Bhi = load16(bp + khalf * 16);        // hi: 16 contiguous ic
        v16bf Blo = load16(bp + 32 + khalf * 16);   // lo
        // bf16x2 split: hi*hi + hi*lo + lo*hi (lo*lo ~2^-32, dropped)
        acc[s] = __builtin_amdgcn_wmma_f32_16x16x32_bf16(
            false, Ah, false, Bhi, (short)0, acc[s], false, false);
        acc[s] = __builtin_amdgcn_wmma_f32_16x16x32_bf16(
            false, Ah, false, Blo, (short)0, acc[s], false, false);
        acc[s] = __builtin_amdgcn_wmma_f32_16x16x32_bf16(
            false, Al, false, Bhi, (short)0, acc[s], false, false);
      }
    }
  }

  // D layout: lane 0-15 -> M = v, lane 16-31 -> M = v+8; N = lane&15.
#pragma unroll
  for (int s = 0; s < 4; ++s) {
    const int px = x0 + s * 16 + n;
#pragma unroll
    for (int v = 0; v < 8; ++v) {
      int m = oc * 16 + v + khalf * 8;
      out[(((size_t)(b * 64 + 32 + m) * HDIM + y) * WDIM) + px] = acc[s][v];
    }
  }
}

// ---------------------------------------------------------------------------
// Motion-blur branch: one thread = one pixel; tap coords/weights computed once
// and reused across all 32 channels (20 gather loads + 20 FMAs per channel).
// ---------------------------------------------------------------------------
__global__ __launch_bounds__(256) void MotionBlur(const float* __restrict__ x,
                                                  const float* __restrict__ mag,
                                                  const float* __restrict__ ori,
                                                  float* __restrict__ out) {
  int pid = blockIdx.x * 256 + threadIdx.x;    // 4*65536 pixels
  int b  = pid >> 16;
  int yx = pid & (HW - 1);
  int y  = yx >> 8;
  int xx = yx & (WDIM - 1);

  float m = mag[pid];   // [B,1,H,W] flattens to pid
  float o = ori[pid];
  float r  = 0.5f * m;
  float dy = r * sinf(o);
  float dx = r * cosf(o);

  int   off[5][4];
  float wgt[5][4];
#pragma unroll
  for (int s = 0; s < 5; ++s) {
    float t  = -1.0f + 0.5f * (float)s;
    float py = (float)y  + t * dy;
    float px = (float)xx + t * dx;
    py = fminf(fmaxf(py, 0.0f), 255.0f);
    px = fminf(fmaxf(px, 0.0f), 255.0f);
    float fy = floorf(py), fx = floorf(px);
    int y0 = (int)fy, x0 = (int)fx;
    int y1 = min(y0 + 1, HDIM - 1);
    int x1 = min(x0 + 1, WDIM - 1);
    float wy = py - fy, wx = px - fx;
    off[s][0] = y0 * WDIM + x0;
    off[s][1] = y0 * WDIM + x1;
    off[s][2] = y1 * WDIM + x0;
    off[s][3] = y1 * WDIM + x1;
    wgt[s][0] = (1.0f - wy) * (1.0f - wx);
    wgt[s][1] = (1.0f - wy) * wx;
    wgt[s][2] = wy * (1.0f - wx);
    wgt[s][3] = wy * wx;
  }

  const float* xb = x + (size_t)b * CIN * HW;
  for (int c = 0; c < CIN; ++c) {
    const float* xc = xb + (size_t)c * HW;
    float acc = 0.0f;
#pragma unroll
    for (int s = 0; s < 5; ++s) {
      acc += wgt[s][0] * xc[off[s][0]];
      acc += wgt[s][1] * xc[off[s][1]];
      acc += wgt[s][2] * xc[off[s][2]];
      acc += wgt[s][3] * xc[off[s][3]];
    }
    out[((size_t)(b * 64 + c) * HW) + yx] = acc * 0.2f;
  }
}

// ---------------------------------------------------------------------------
extern "C" void kernel_launch(void* const* d_in, const int* in_sizes, int n_in,
                              void* d_out, int out_size, void* d_ws, size_t ws_size,
                              hipStream_t stream) {
  const float* x   = (const float*)d_in[0];   // [4,32,256,256]
  const float* mag = (const float*)d_in[1];   // [4,1,256,256]
  const float* ori = (const float*)d_in[2];   // [4,1,256,256]
  const float* w   = (const float*)d_in[3];   // [32,32,5,5]
  float* out = (float*)d_out;                 // [4,64,256,256]

  bf16_t* whi = (bf16_t*)d_ws;                // 25600 bf16 = 50KB
  bf16_t* wlo = whi + 25 * 2 * 32 * 16;       // +50KB  (total 100KB of ws)

  PrepWeightFrag<<<100, 256, 0, stream>>>(w, whi, wlo);
  MotionBlur<<<(4 * HW) / 256, 256, 0, stream>>>(x, mag, ori, out);
  ConvWmma<<<dim3(4, HDIM, 4), 64, 0, stream>>>(x, whi, wlo, out);
}
